// SetTransformer_8701603741928
// MI455X (gfx1250) — matrix operand order repaired and checked by reading, not compiled, not run on hardware
//
#include <hip/hip_runtime.h>
#include <math.h>

// ---------------------------------------------------------------------------
// Set Transformer forward on gfx1250 (MI455X).
//  - Projection / rFF / output GEMMs: V_WMMA_F32_16X16X32_BF16, bf16x3 split
//    (hi+lo) for ~fp32 accuracy; weights staged to LDS with
//    GLOBAL_LOAD_ASYNC_TO_LDS_B128 (ASYNCcnt double-buffering), shared by the
//    4 waves of a workgroup.
//  - Attention (QK^T, P*V) + residual: exact fp32 via V_WMMA_F32_16X16X4_F32,
//    K/V tiles staged via async global->LDS copies.
// ---------------------------------------------------------------------------

typedef __attribute__((ext_vector_type(2)))  float  v2f;
typedef __attribute__((ext_vector_type(8)))  float  v8f;
typedef __attribute__((ext_vector_type(16))) __bf16 v16bf;
typedef unsigned short us16;

#define WMMA_F32(a, b, c) \
  __builtin_amdgcn_wmma_f32_16x16x4_f32(false, (a), false, (b), (short)0, (c), false, false)
#define WMMA_BF16(a, b, c) \
  __builtin_amdgcn_wmma_f32_16x16x32_bf16(false, (a), false, (b), (short)0, (c), false, false)

#define DV 1024
#define NH 8
#define DH 128
#define NB 16

// --------------------------- async copy helpers ----------------------------
// CDNA5: the LDS address is taken from the VDST VGPR; flat-pointer low 32 bits are the
// logical LDS address (ISA 10.2), so we pass those as the LDS operand.
__device__ __forceinline__ void async_b128(void* lds, const void* g) {
  asm volatile("global_load_async_to_lds_b128 %0, %1, off"
               :: "v"((unsigned)(unsigned long long)lds), "v"(g) : "memory");
}
__device__ __forceinline__ void wait_async0() {
  asm volatile("s_wait_asynccnt 0x0" ::: "memory");
}
__device__ __forceinline__ void wait_async4() {
  asm volatile("s_wait_asynccnt 0x4" ::: "memory");
}
__device__ __forceinline__ void wait_ds0() {
  asm volatile("s_wait_dscnt 0x0" ::: "memory");
}

// ------------------------------ bf16 helpers -------------------------------
__device__ __forceinline__ us16 f2bf(float x) {
  unsigned u = __builtin_bit_cast(unsigned int, x);
  u += 0x7FFFu + ((u >> 16) & 1u);          // round-to-nearest-even
  return (us16)(u >> 16);
}
__device__ __forceinline__ float bf2f(us16 h) {
  unsigned u = ((unsigned)h) << 16;
  return __builtin_bit_cast(float, u);
}

// ---------------------------------------------------------------------------
// Split f32 -> (bf16 hi, bf16 lo) elementwise.  lo = bf16(x - float(hi)).
// ---------------------------------------------------------------------------
__global__ __launch_bounds__(256)
void split_kernel(const float* __restrict__ x, us16* __restrict__ hi,
                  us16* __restrict__ lo, long n)
{
  long i = (long)blockIdx.x * 256 + threadIdx.x;
  const long step = (long)gridDim.x * 256;
  for (; i < n; i += step) {
    const float v = x[i];
    const us16 h = f2bf(v);
    hi[i] = h;
    lo[i] = f2bf(v - bf2f(h));
  }
}

// ---------------------------------------------------------------------------
// Repack weight W[K,N] (f32) into WMMA-B fragment-major bf16 hi/lo:
// dword ((kc*(N/16)+nt)*256 + lane*8 + v) packs rows {kc*32+half*16+2v, +1}
// at column nt*16 + (lane&15).
// ---------------------------------------------------------------------------
__global__ __launch_bounds__(256)
void repack_w_kernel(const float* __restrict__ W, int K, int N,
                     unsigned* __restrict__ Whi, unsigned* __restrict__ Wlo)
{
  const int T = (K >> 5) * (N >> 4) * 32;
  const int t = blockIdx.x * 256 + threadIdx.x;
  if (t >= T) return;
  const int l    = t & 31;
  const int tile = t >> 5;                 // kc * (N/16) + nt
  const int ntiles = N >> 4;
  const int kc = tile / ntiles;
  const int nt = tile - kc * ntiles;
  const int lm = l & 15, lh = l >> 4;
  const int n  = nt * 16 + lm;
  const long base = (long)tile * 256 + l * 8;
#pragma unroll
  for (int v = 0; v < 8; ++v) {
    const int k = kc * 32 + lh * 16 + 2 * v;
    const float w0 = W[(long)k * N + n];
    const float w1 = W[(long)(k + 1) * N + n];
    const us16 h0 = f2bf(w0), h1 = f2bf(w1);
    const us16 e0 = f2bf(w0 - bf2f(h0)), e1 = f2bf(w1 - bf2f(h1));
    Whi[base + v] = (unsigned)h0 | ((unsigned)h1 << 16);
    Wlo[base + v] = (unsigned)e0 | ((unsigned)e1 << 16);
  }
}

// ---------------------------------------------------------------------------
// Split-precision bf16 GEMM + bias, 4 waves/block, W staged to LDS async.
//   Y[b,m,n] = sum_k A[b,m,k]*W[k,n] + bias[n];  C += Ah*Wh + Ah*Wl + Al*Wh.
// Wave w owns M-tile blockIdx.x*64 + w*16; all waves share 64 N-columns.
// grid = (ceil(M/64), N/64, B), block = 128.
// ---------------------------------------------------------------------------
union FragB { uint4 q[2]; v16bf v; };

__global__ __launch_bounds__(128)
void gemm_bf16_kernel(const us16* __restrict__ Ahi, const us16* __restrict__ Alo,
                      long saB, int M, int K,
                      const unsigned* __restrict__ Whi, const unsigned* __restrict__ Wlo,
                      const float* __restrict__ bias, float* __restrict__ Y, int N)
{
  __shared__ unsigned Wlds[2 * 2 * 1024];   // [buf][hi/lo][4 tiles * 256 dwords]

  const int tid  = threadIdx.x;
  const int l    = tid & 31;
  const int wave = tid >> 5;
  const int lm = l & 15;
  const int lh = l >> 4;
  const int m0 = blockIdx.x * 64 + wave * 16;
  const int n0 = blockIdx.y * 64;
  const int b  = blockIdx.z;
  const int ntiles = N >> 4;
  const int nt0 = n0 >> 4;

  int arow = m0 + lm; if (arow >= M) arow = M - 1;
  const us16* arh = Ahi + (long)b * saB + (long)arow * K;
  const us16* arl = Alo + (long)b * saB + (long)arow * K;

  v8f c0 = {0,0,0,0,0,0,0,0};
  v8f c1 = {0,0,0,0,0,0,0,0};
  v8f c2 = {0,0,0,0,0,0,0,0};
  v8f c3 = {0,0,0,0,0,0,0,0};

  const int nkc = K >> 5;

  // async-stage packed W chunk kc into LDS buffer d (4 x b128 per thread)
  auto issueW = [&](int kc, int d) {
    const unsigned* sH = Whi + ((long)kc * ntiles + nt0) * 256 + tid * 8;
    const unsigned* sL = Wlo + ((long)kc * ntiles + nt0) * 256 + tid * 8;
    unsigned* dH = &Wlds[(d * 2 + 0) * 1024 + tid * 8];
    unsigned* dL = &Wlds[(d * 2 + 1) * 1024 + tid * 8];
    async_b128(dH,     sH);
    async_b128(dH + 4, sH + 4);
    async_b128(dL,     sL);
    async_b128(dL + 4, sL + 4);
  };

  issueW(0, 0);
  for (int kc = 0; kc < nkc; ++kc) {
    const int d = kc & 1;
    if (kc + 1 < nkc) { issueW(kc + 1, d ^ 1); wait_async4(); }
    else              { wait_async0(); }
    __syncthreads();

    const int k0 = kc * 32;
    FragB ah, al;
    ah.q[0] = *(const uint4*)(arh + k0 + lh * 16);
    ah.q[1] = *(const uint4*)(arh + k0 + lh * 16 + 8);
    al.q[0] = *(const uint4*)(arl + k0 + lh * 16);
    al.q[1] = *(const uint4*)(arl + k0 + lh * 16 + 8);
    if (kc + 1 < nkc) {                      // -> global_prefetch_b8
      __builtin_prefetch(arh + k0 + 32, 0, 1);
      __builtin_prefetch(arl + k0 + 32, 0, 1);
    }

    const unsigned* WH = &Wlds[(d * 2 + 0) * 1024];
    const unsigned* WL = &Wlds[(d * 2 + 1) * 1024];
#pragma unroll
    for (int j = 0; j < 4; ++j) {
      FragB wh, wl;
      wh.q[0] = *(const uint4*)(WH + j * 256 + l * 8);
      wh.q[1] = *(const uint4*)(WH + j * 256 + l * 8 + 4);
      wl.q[0] = *(const uint4*)(WL + j * 256 + l * 8);
      wl.q[1] = *(const uint4*)(WL + j * 256 + l * 8 + 4);
      v8f* cj = (j == 0) ? &c0 : (j == 1) ? &c1 : (j == 2) ? &c2 : &c3;
      *cj = WMMA_BF16(ah.v, wh.v, *cj);
      *cj = WMMA_BF16(ah.v, wl.v, *cj);
      *cj = WMMA_BF16(al.v, wh.v, *cj);
    }
    __syncthreads();   // all waves done reading buf d before it is re-filled
  }

  float* Yb = Y + (long)b * (long)M * N;
  const float bv0 = bias[n0 + lm];
  const float bv1 = bias[n0 + 16 + lm];
  const float bv2 = bias[n0 + 32 + lm];
  const float bv3 = bias[n0 + 48 + lm];
#pragma unroll
  for (int v = 0; v < 8; ++v) {
    const int row = m0 + v + lh * 8;
    if (row < M) {
      float* yp = Yb + (long)row * N + n0 + lm;
      yp[0]  = c0[v] + bv0;
      yp[16] = c1[v] + bv1;
      yp[32] = c2[v] + bv2;
      yp[48] = c3[v] + bv3;
    }
  }
}

// ---------------------------------------------------------------------------
// Multi-head attention + Q residual, exact fp32 WMMA (16x16x4), flash-style.
// One wave per (batch, head, 16-query tile); K/V tiles async-staged to LDS.
// ---------------------------------------------------------------------------
#define QSTR 132

__global__ __launch_bounds__(32)
void attn_kernel(float* __restrict__ QO, const float* __restrict__ Kp,
                 const float* __restrict__ Vp, int nq, int nk)
{
  __shared__ float Qs[16 * QSTR];
  __shared__ float Ks[16 * QSTR];
  __shared__ float Vs[16 * QSTR];
  __shared__ float Ps[16 * 16];

  const int l  = threadIdx.x;
  const int lm = l & 15;
  const int lh = l >> 4;
  const int h  = blockIdx.y;
  const int b  = blockIdx.z;
  const int m0 = blockIdx.x * 16;

  const float* Qb = QO + ((long)b * nq) * DV + h * DH;
  const float* Kb = Kp + ((long)b * nk) * DV + h * DH;
  const float* Vb = Vp + ((long)b * nk) * DV + h * DH;

  for (int idx = l; idx < 16 * 32; idx += 32) {   // async-stage Q tile
    const int r = idx >> 5, c4 = (idx & 31) << 2;
    int row = m0 + r; if (row >= nq) row = nq - 1;
    async_b128(&Qs[r * QSTR + c4], Qb + (long)row * DV + c4);
  }
  wait_async0();
  __syncthreads();

  float m_i[8], l_i[8];
  v8f acc[8];
  const v8f z8 = {0,0,0,0,0,0,0,0};
#pragma unroll
  for (int v = 0; v < 8; ++v) { m_i[v] = -__builtin_inff(); l_i[v] = 0.f; acc[v] = z8; }

  const int nchunks = (nk + 15) >> 4;
  for (int kc = 0; kc < nchunks; ++kc) {
    wait_ds0();    // WAR: async LDS writes are unordered vs prior DS reads
    for (int idx = l; idx < 16 * 32; idx += 32) {
      const int r = idx >> 5, c4 = (idx & 31) << 2;
      int row = kc * 16 + r; if (row >= nk) row = nk - 1;
      async_b128(&Ks[r * QSTR + c4], Kb + (long)row * DV + c4);
      async_b128(&Vs[r * QSTR + c4], Vb + (long)row * DV + c4);
    }
    wait_async0();
    __syncthreads();

    v8f s = z8;                                  // S = Qh Kh^T (32 WMMAs)
#pragma unroll
    for (int k0 = 0; k0 < DH; k0 += 4) {
      const int ko = k0 + lh * 2;
      v2f a;  a.x  = Qs[lm * QSTR + ko]; a.y  = Qs[lm * QSTR + ko + 1];
      v2f bb; bb.x = Ks[lm * QSTR + ko]; bb.y = Ks[lm * QSTR + ko + 1];
      s = WMMA_F32(a, bb, s);
    }

    const bool valid = (kc * 16 + lm) < nk;
#pragma unroll
    for (int v = 0; v < 8; ++v)
      s[v] = valid ? s[v] * 0.03125f : -__builtin_inff();   // 1/sqrt(1024)

#pragma unroll
    for (int v = 0; v < 8; ++v) {                // online softmax, row v+lh*8
      float t = s[v];
      for (int off = 8; off; off >>= 1) t = fmaxf(t, __shfl_xor(t, off, 16));
      const float mnew = fmaxf(m_i[v], t);
      const float pe = __expf(s[v] - mnew);
      float ps = pe;
      for (int off = 8; off; off >>= 1) ps += __shfl_xor(ps, off, 16);
      const float sc = __expf(m_i[v] - mnew);
      l_i[v] = l_i[v] * sc + ps;
      m_i[v] = mnew;
#pragma unroll
      for (int j = 0; j < 8; ++j) acc[j][v] *= sc;
      Ps[(v + lh * 8) * 16 + lm] = pe;
    }
    __syncthreads();

#pragma unroll
    for (int j = 0; j < 8; ++j) {                // acc += P * Vchunk (32 WMMAs)
      const int col = j * 16 + lm;
#pragma unroll
      for (int k0 = 0; k0 < 16; k0 += 4) {
        const int ko = k0 + lh * 2;
        v2f a;  a.x  = Ps[lm * 16 + ko];     a.y  = Ps[lm * 16 + ko + 1];
        v2f bb; bb.x = Vs[ko * QSTR + col];  bb.y = Vs[(ko + 1) * QSTR + col];
        acc[j] = WMMA_F32(a, bb, acc[j]);
      }
    }
    __syncthreads();
  }

#pragma unroll
  for (int j = 0; j < 8; ++j)
#pragma unroll
    for (int v = 0; v < 8; ++v) {
      const int r = v + lh * 8;
      const int row = m0 + r;
      if (row < nq) {
        const float val = acc[j][v] / l_i[v] + Qs[r * QSTR + j * 16 + lm];
        QO[((long)b * nq + row) * DV + h * DH + j * 16 + lm] = val;
      }
    }
}

// ---------------------------------------------------------------------------
// LayerNorm over 1024 features; optional fused rFF residual t = x + relu(G).
// ---------------------------------------------------------------------------
__global__ __launch_bounds__(256)
void ln_kernel(const float* __restrict__ x, const float* __restrict__ gmat,
               const float* __restrict__ g, const float* __restrict__ be,
               float* __restrict__ dst)
{
  __shared__ float red[256];
  const long row = blockIdx.x;
  const float* xr = x + row * DV;
  float* dr = dst + row * DV;
  const int t = threadIdx.x;

  float vals[4];
  float s = 0.f;
#pragma unroll
  for (int i = 0; i < 4; ++i) {
    const int c = t + i * 256;
    float v = xr[c];
    if (gmat) v += fmaxf(gmat[row * DV + c], 0.f);
    vals[i] = v;
    s += v;
  }
  red[t] = s; __syncthreads();
  for (int o = 128; o; o >>= 1) { if (t < o) red[t] += red[t + o]; __syncthreads(); }
  const float mean = red[0] * (1.f / DV);
  __syncthreads();

  float ss = 0.f;
#pragma unroll
  for (int i = 0; i < 4; ++i) { const float d = vals[i] - mean; ss += d * d; }
  red[t] = ss; __syncthreads();
  for (int o = 128; o; o >>= 1) { if (t < o) red[t] += red[t + o]; __syncthreads(); }
  const float rstd = rsqrtf(red[0] * (1.f / DV) + 1e-5f);

#pragma unroll
  for (int i = 0; i < 4; ++i) {
    const int c = t + i * 256;
    dr[c] = (vals[i] - mean) * rstd * g[c] + be[c];
  }
}

// ---------------------------------------------------------------------------
// Host orchestration
// ---------------------------------------------------------------------------
struct Mab {  // JAX sorted-key order: Wk,Wo,Wq,Wv,be0,be1,bk,bo,bq,bv,g0,g1
  const float *Wk, *Wo, *Wq, *Wv, *be0, *be1, *bk, *bo, *bq, *bv, *g0, *g1;
};
static Mab mab_at(void* const* d, int i) {
  Mab m;
  m.Wk  = (const float*)d[i + 0];  m.Wo  = (const float*)d[i + 1];
  m.Wq  = (const float*)d[i + 2];  m.Wv  = (const float*)d[i + 3];
  m.be0 = (const float*)d[i + 4];  m.be1 = (const float*)d[i + 5];
  m.bk  = (const float*)d[i + 6];  m.bo  = (const float*)d[i + 7];
  m.bq  = (const float*)d[i + 8];  m.bv  = (const float*)d[i + 9];
  m.g0  = (const float*)d[i + 10]; m.g1  = (const float*)d[i + 11];
  return m;
}

struct Scratch {
  float *bufA, *bufB, *Qp, *Kp, *Vp, *bufH;
  us16  *QinH, *QinL, *KinH, *KinL;
  unsigned *WqH, *WqL, *WkH, *WkL, *WvH, *WvL, *WoH, *WoL, *WoutH, *WoutL;
};

static inline int split_grid(long n) {
  long b = (n + 255) / 256;
  return (int)(b > 8192 ? 8192 : b);
}

static void run_mab(hipStream_t s, const Mab& p,
                    const float* Qf, long sQ, int nq, int dq,
                    const float* Kf, long sK, int nk, int dk,
                    float* out, const Scratch& sc)
{
  const int qt = (nq + 15) / 16, kt = (nk + 15) / 16;
  const int qg = (nq + 63) / 64, kg = (nk + 63) / 64;   // 4-wave GEMM blocks
  const long nQ = (sQ ? (long)NB * nq * dq : (long)nq * dq);
  const long nK = (sK ? (long)NB * nk * dk : (long)nk * dk);
  const long sQb = sQ ? (long)nq * dq : 0;
  const long sKb = sK ? (long)nk * dk : 0;

  split_kernel<<<split_grid(nQ), 256, 0, s>>>(Qf, sc.QinH, sc.QinL, nQ);
  split_kernel<<<split_grid(nK), 256, 0, s>>>(Kf, sc.KinH, sc.KinL, nK);
  repack_w_kernel<<<((dq / 32) * (DV / 16) * 32 + 255) / 256, 256, 0, s>>>(p.Wq, dq, DV, sc.WqH, sc.WqL);
  repack_w_kernel<<<((dk / 32) * (DV / 16) * 32 + 255) / 256, 256, 0, s>>>(p.Wk, dk, DV, sc.WkH, sc.WkL);
  repack_w_kernel<<<((dk / 32) * (DV / 16) * 32 + 255) / 256, 256, 0, s>>>(p.Wv, dk, DV, sc.WvH, sc.WvL);
  repack_w_kernel<<<((DV / 32) * (DV / 16) * 32 + 255) / 256, 256, 0, s>>>(p.Wo, DV, DV, sc.WoH, sc.WoL);

  gemm_bf16_kernel<<<dim3(qg, DV / 64, NB), 128, 0, s>>>(sc.QinH, sc.QinL, sQb, nq, dq, sc.WqH, sc.WqL, p.bq, sc.Qp, DV);
  gemm_bf16_kernel<<<dim3(kg, DV / 64, NB), 128, 0, s>>>(sc.KinH, sc.KinL, sKb, nk, dk, sc.WkH, sc.WkL, p.bk, sc.Kp, DV);
  gemm_bf16_kernel<<<dim3(kg, DV / 64, NB), 128, 0, s>>>(sc.KinH, sc.KinL, sKb, nk, dk, sc.WvH, sc.WvL, p.bv, sc.Vp, DV);

  attn_kernel<<<dim3(qt, NH, NB), 32, 0, s>>>(sc.Qp, sc.Kp, sc.Vp, nq, nk);
  ln_kernel<<<NB * nq, 256, 0, s>>>(sc.Qp, nullptr, p.g0, p.be0, sc.Qp);

  split_kernel<<<split_grid((long)NB * nq * DV), 256, 0, s>>>(sc.Qp, sc.QinH, sc.QinL, (long)NB * nq * DV);
  gemm_bf16_kernel<<<dim3(qg, DV / 64, NB), 128, 0, s>>>(sc.QinH, sc.QinL, (long)nq * DV, nq, DV, sc.WoH, sc.WoL, p.bo, sc.Kp, DV);
  ln_kernel<<<NB * nq, 256, 0, s>>>(sc.Qp, sc.Kp, p.g1, p.be1, out);
}

extern "C" void kernel_launch(void* const* d_in, const int* in_sizes, int n_in,
                              void* d_out, int out_size, void* d_ws, size_t ws_size,
                              hipStream_t stream)
{
  // d_in: X, then params as JAX pytree (dict keys sorted):
  //   S, Wout, bout, enc0{I,mab0,mab1}, enc1{...}, pma, sab1, sab2
  const float* X    = (const float*)d_in[0];
  const float* S    = (const float*)d_in[1];
  const float* Wout = (const float*)d_in[2];
  const float* bout = (const float*)d_in[3];
  const float* I0   = (const float*)d_in[4];
  const Mab e0m0 = mab_at(d_in, 5);
  const Mab e0m1 = mab_at(d_in, 17);
  const float* I1   = (const float*)d_in[29];
  const Mab e1m0 = mab_at(d_in, 30);
  const Mab e1m1 = mab_at(d_in, 42);
  const Mab pma  = mab_at(d_in, 54);
  const Mab sab1 = mab_at(d_in, 66);
  const Mab sab2 = mab_at(d_in, 78);

  const long S1 = (long)NB * 2048 * DV;     // 33,554,432 elements
  char* w = (char*)d_ws;
  auto take = [&](size_t bytes) { char* p = w; w += (bytes + 255) & ~(size_t)255; return p; };

  Scratch sc;
  sc.bufA = (float*)take(S1 * 4);
  sc.bufB = (float*)take(S1 * 4);
  sc.Qp   = (float*)take(S1 * 4);
  sc.Kp   = (float*)take(S1 * 4);
  sc.Vp   = (float*)take(S1 * 4);
  sc.bufH = (float*)take((size_t)NB * 64 * DV * 4);
  sc.QinH = (us16*)take(S1 * 2);  sc.QinL = (us16*)take(S1 * 2);
  sc.KinH = (us16*)take(S1 * 2);  sc.KinL = (us16*)take(S1 * 2);
  const size_t WPB = (size_t)DV * DV / 2 * 4;   // max packed-weight bytes
  sc.WqH = (unsigned*)take(WPB); sc.WqL = (unsigned*)take(WPB);
  sc.WkH = (unsigned*)take(WPB); sc.WkL = (unsigned*)take(WPB);
  sc.WvH = (unsigned*)take(WPB); sc.WvL = (unsigned*)take(WPB);
  sc.WoH = (unsigned*)take(WPB); sc.WoL = (unsigned*)take(WPB);
  sc.WoutH = (unsigned*)take(WPB); sc.WoutL = (unsigned*)take(WPB);

  // enc0 (ISAB): H = MAB(I0, X);  X1 = MAB(X, H)
  run_mab(stream, e0m0, I0, 0,           64,   DV,  X,       2048L * 256, 2048, 256, sc.bufH, sc);
  run_mab(stream, e0m1, X,  2048L * 256, 2048, 256, sc.bufH, 64L * DV,    64,   DV,  sc.bufA, sc);
  // enc1 (ISAB): H = MAB(I1, X1); X2 = MAB(X1, H)
  run_mab(stream, e1m0, I1,      0,          64,   DV, sc.bufA, 2048L * DV, 2048, DV, sc.bufH, sc);
  run_mab(stream, e1m1, sc.bufA, 2048L * DV, 2048, DV, sc.bufH, 64L * DV,   64,   DV, sc.bufB, sc);
  // PMA: X3 = MAB(S, X2) -> [B,1,1024]
  run_mab(stream, pma,  S,       0,          1,    DV, sc.bufB, 2048L * DV, 2048, DV, sc.bufA, sc);
  // SAB1, SAB2 on the single seed
  run_mab(stream, sab1, sc.bufA, (long)DV,   1,    DV, sc.bufA, (long)DV,   1,    DV, sc.bufB, sc);
  run_mab(stream, sab2, sc.bufB, (long)DV,   1,    DV, sc.bufB, (long)DV,   1,    DV, sc.bufA, sc);
  // output projection: [B,1,1024] @ [1024,256] + bout -> d_out
  split_kernel<<<split_grid((long)NB * DV), 256, 0, stream>>>(sc.bufA, sc.QinH, sc.QinL, (long)NB * DV);
  repack_w_kernel<<<((DV / 32) * (256 / 16) * 32 + 255) / 256, 256, 0, stream>>>(Wout, DV, 256, sc.WoutH, sc.WoutL);
  gemm_bf16_kernel<<<dim3(1, 256 / 64, NB), 128, 0, stream>>>(
      sc.QinH, sc.QinL, (long)DV, 1, DV, sc.WoutH, sc.WoutL, bout, (float*)d_out, 256);
}
